// CausalSelfAttention_16003048145608
// MI455X (gfx1250) — compile-verified, matrix operand order checked by reading, not run on hardware
//
#include <hip/hip_runtime.h>

// ---------------------------------------------------------------------------
// CDNA5 (gfx1250) causal self-attention, bf16 WMMA pipeline + TDM async K/V
// staging. B=4, T=1024, C=2048, H=16, HD=128, BLOCK=1024. All matmuls via
// v_wmma_f32_16x16x32_bf16 (fp32 accumulate).
// ---------------------------------------------------------------------------

typedef __attribute__((ext_vector_type(16))) __bf16         v16bf;
typedef __attribute__((ext_vector_type(16))) unsigned short v16us;
typedef __attribute__((ext_vector_type(8)))  float          v8f;
typedef __attribute__((ext_vector_type(4)))  unsigned int   v4u;
typedef __attribute__((ext_vector_type(8)))  unsigned int   v8u;

__device__ __forceinline__ unsigned short f2bf(float f) {
  unsigned u = __builtin_bit_cast(unsigned, f);
  unsigned r = u + 0x7fffu + ((u >> 16) & 1u);   // round-to-nearest-even
  return (unsigned short)(r >> 16);
}
__device__ __forceinline__ float bf2f(unsigned short h) {
  unsigned u = ((unsigned)h) << 16;
  return __builtin_bit_cast(float, u);
}
__device__ __forceinline__ v8f wmma_bf16(v16us a, v16us b, v8f c) {
  return __builtin_amdgcn_wmma_f32_16x16x32_bf16(
      false, __builtin_bit_cast(v16bf, a),
      false, __builtin_bit_cast(v16bf, b),
      (short)0, c, false, false);
}
// 16-bit A/B fragment K-index map (ISA 7.12.2): lanes 0-15 hold K 0-7,16-23;
// lanes 16-31 hold K 8-15,24-31 (2 halves per VGPR).
__device__ __forceinline__ int kmap(int i, int g) {
  return (i & 7) + ((i >> 3) << 4) + (g << 3);
}

// ---------------------------------------------------------------------------
// Tensor Data Mover: DMA a 64-row x 128-col bf16 tile (row stride 128 elems)
// from global memory into LDS, padding each 256B row with 16B so the LDS view
// is [64][136] shorts (bank-conflict-free columns). D# per ISA ch.8:
//   group0: count=1, lds_addr, global_addr[56:0], type=2
//   group1: data_size=2B, pad_enable, pad_interval=64 DW, pad_amount=4 DW,
//           tensor_dim0=128, tensor_dim1=1024, tile_dim0=128, tile_dim1=64,
//           tensor_dim0_stride=128, workgroup_mask=0 (not in cluster)
// Tracked by TENSORcnt; VADDR2/3 omitted (NULL) for a 2D tensor.
// ---------------------------------------------------------------------------
__device__ __forceinline__ void tdm_load_tile(unsigned lds_off,
                                              const unsigned short* g) {
  unsigned long long ga = (unsigned long long)(size_t)g;
  v4u g0;
  g0[0] = 1u;                                    // count=1, user descriptor
  g0[1] = lds_off;                               // LDS byte address
  g0[2] = (unsigned)ga;                          // global_addr[31:0]
  g0[3] = (unsigned)(ga >> 32) | (2u << 30);     // global_addr[56:32] | type=2
  v8u g1;
  g1[0] = 0x07510000u;   // data_size=1(2B) | pad_enable | interval=5 | amount=3
  g1[1] = 128u << 16;    // tensor_dim0[15:0]=128  (bits 79:48 -> dw1[31:16])
  g1[2] = 1024u << 16;   // tensor_dim1[15:0]=1024 (bits 111:80 -> dw2[31:16])
  g1[3] = 128u << 16;    // tile_dim0=128          (bits 127:112)
  g1[4] = 64u;           // tile_dim1=64           (bits 143:128)
  g1[5] = 128u;          // tensor_dim0_stride=128 (bits 207:160)
  g1[6] = 0u;
  g1[7] = 0u;
  asm volatile("tensor_load_to_lds %0, %1" :: "s"(g0), "s"(g1) : "memory");
}

// ---------------------------------------------------------------------------
// Kernel 1: counts histogram (over ALL batches, matching the JAX scatter),
// reciprocal, inclusive scan over T, then sin/cos tables (B,T,64).
// ---------------------------------------------------------------------------
__global__ __launch_bounds__(1024)
void sincos_kernel(const int* __restrict__ token_index,
                   float* __restrict__ sinb, float* __restrict__ cosb) {
  __shared__ int   cnt[1024];
  __shared__ float sc[1024];
  const int tid = threadIdx.x, b = blockIdx.x;
  cnt[tid] = 0;
  __syncthreads();
  #pragma unroll
  for (int bb = 0; bb < 4; ++bb)
    atomicAdd(&cnt[token_index[bb * 1024 + tid]], 1);
  __syncthreads();
  const int myi = token_index[b * 1024 + tid];
  sc[tid] = 1.0f / ((float)cnt[myi] + 1e-10f);
  __syncthreads();
  for (int off = 1; off < 1024; off <<= 1) {      // Hillis-Steele inclusive scan
    float add = (tid >= off) ? sc[tid - off] : 0.0f;
    __syncthreads();
    sc[tid] += add;
    __syncthreads();
  }
  const float tval = sc[tid];
  for (int j = 0; j < 64; ++j) {
    float f = tval * __powf(10000.0f, -(float)j * (1.0f / 64.0f));
    int o = (b * 1024 + tid) * 64 + j;
    sinb[o] = __sinf(f);
    cosb[o] = __cosf(f);
  }
}

// ---------------------------------------------------------------------------
// Kernel 2/5: tiled bf16 WMMA GEMM. 128x128 block tile, BK=32, 8 waves.
// MODE 0: f32 row-major output. MODE 1: scatter bf16 into head-layout Q/K/V.
// AT = float (convert to bf16 in staging) or unsigned short (bf16 passthrough).
// ---------------------------------------------------------------------------
template <int MODE, typename AT>
__global__ __launch_bounds__(256)
void gemm_wmma(const AT* __restrict__ A, const float* __restrict__ Bm,
               float* __restrict__ Cout,
               unsigned short* __restrict__ Qo, unsigned short* __restrict__ Ko,
               unsigned short* __restrict__ Vo, int M, int N, int K) {
  __shared__ unsigned short As[128][40];    // 128x32 + pad
  __shared__ unsigned short Bs[32][136];    // 32x128 + pad
  const int tid = threadIdx.x;
  const int lane = tid & 31, wid = tid >> 5;
  const int g = lane >> 4, lm = lane & 15;
  const int wm = wid & 3, wn = wid >> 2;            // 4x2 wave grid
  const int bM = blockIdx.y * 128, bN = blockIdx.x * 128;

  v8f acc[2][4];
  #pragma unroll
  for (int i = 0; i < 2; ++i)
    #pragma unroll
    for (int j = 0; j < 4; ++j)
      #pragma unroll
      for (int e = 0; e < 8; ++e) acc[i][j][e] = 0.0f;

  const int arow = tid >> 1, acb = (tid & 1) * 16;  // A staging: 16 elems/thread
  const int brow = tid >> 3, bcb = (tid & 7) * 16;  // B staging: 16 elems/thread

  for (int k0 = 0; k0 < K; k0 += 32) {
    if constexpr (sizeof(AT) == 4) {
      const float* src = (const float*)A + (size_t)(bM + arow) * K + k0 + acb;
      if (k0 + 32 < K) __builtin_prefetch(src + 32, 0, 0);   // global_prefetch_b8
      #pragma unroll
      for (int i = 0; i < 16; ++i) As[arow][acb + i] = f2bf(src[i]);
    } else {
      const unsigned short* src =
          (const unsigned short*)A + (size_t)(bM + arow) * K + k0 + acb;
      if (k0 + 32 < K) __builtin_prefetch(src + 32, 0, 0);
      #pragma unroll
      for (int i = 0; i < 16; ++i) As[arow][acb + i] = src[i];
    }
    {
      const float* src = Bm + (size_t)(k0 + brow) * N + bN + bcb;
      if (k0 + 32 < K) __builtin_prefetch(src + (size_t)32 * N, 0, 0);
      #pragma unroll
      for (int i = 0; i < 16; ++i) Bs[brow][bcb + i] = f2bf(src[i]);
    }
    __syncthreads();

    v16us afrag[2], bfrag[4];
    #pragma unroll
    for (int sm = 0; sm < 2; ++sm) {
      const int m = wm * 32 + sm * 16 + lm;
      #pragma unroll
      for (int i = 0; i < 16; ++i) afrag[sm][i] = As[m][kmap(i, g)];
    }
    #pragma unroll
    for (int sn = 0; sn < 4; ++sn) {
      const int n = wn * 64 + sn * 16 + lm;
      #pragma unroll
      for (int i = 0; i < 16; ++i) bfrag[sn][i] = Bs[kmap(i, g)][n];
    }
    #pragma unroll
    for (int sm = 0; sm < 2; ++sm)
      #pragma unroll
      for (int sn = 0; sn < 4; ++sn)
        acc[sm][sn] = wmma_bf16(afrag[sm], bfrag[sn], acc[sm][sn]);
    __syncthreads();
  }

  #pragma unroll
  for (int sm = 0; sm < 2; ++sm)
    #pragma unroll
    for (int sn = 0; sn < 4; ++sn)
      #pragma unroll
      for (int e = 0; e < 8; ++e) {
        const int row = bM + wm * 32 + sm * 16 + g * 8 + e;
        const int col = bN + wn * 64 + sn * 16 + lm;
        const float val = acc[sm][sn][e];
        if constexpr (MODE == 0) {
          Cout[(size_t)row * N + col] = val;
        } else {
          // row = b*1024 + t ; col = which*2048 + h*128 + d
          const int b = row >> 10, t = row & 1023;
          const int which = col >> 11, c = col & 2047;
          const int h = c >> 7, d = c & 127;
          const size_t dst = ((size_t)((b * 16 + h) * 1024 + t)) * 128 + d;
          const unsigned short bv = f2bf(val);
          if (which == 0)      Qo[dst] = bv;
          else if (which == 1) Ko[dst] = bv;
          else                 Vo[dst] = bv;
        }
      }
}

// ---------------------------------------------------------------------------
// Kernel 3: rotary embedding on Q/K (in place, bf16), last-element overrides,
// V scaled by exp(cumulative_scores). One thread per (b,h,t,j<64) pair.
// ---------------------------------------------------------------------------
__global__ __launch_bounds__(256)
void rope_kernel(unsigned short* __restrict__ Q, unsigned short* __restrict__ Kb,
                 unsigned short* __restrict__ Vb,
                 const float* __restrict__ sinb, const float* __restrict__ cosb,
                 const float* __restrict__ cs) {
  const int idx = blockIdx.x * 256 + threadIdx.x;   // B*H*T*64 total
  const int j = idx & 63;
  const int t = (idx >> 6) & 1023;
  const int h = (idx >> 16) & 15;
  const int b = idx >> 20;
  const size_t base = ((size_t)((b * 16 + h) * 1024 + t)) * 128;
  const int sci = (b * 1024 + t) * 64 + j;
  const float sn = sinb[sci], cn = cosb[sci];

  const float q1 = bf2f(Q[base + j]), q2 = bf2f(Q[base + j + 64]);
  float nq1 = q1 * cn - q2 * sn;
  float nq2 = q2 * cn + q1 * sn;
  const float k1 = bf2f(Kb[base + j]), k2 = bf2f(Kb[base + j + 64]);
  float nk1 = k1 * cn - k2 * sn;
  float nk2 = k2 * cn + k1 * sn;
  const float c = cs[b * 1024 + t];
  if (j == 63) { nq2 = 1.0f; nk2 = c; }
  Q[base + j] = f2bf(nq1);  Q[base + j + 64] = f2bf(nq2);
  Kb[base + j] = f2bf(nk1); Kb[base + j + 64] = f2bf(nk2);
  const float vsc = __expf(c);
  Vb[base + j]      = f2bf(bf2f(Vb[base + j]) * vsc);
  Vb[base + j + 64] = f2bf(bf2f(Vb[base + j + 64]) * vsc);
}

// ---------------------------------------------------------------------------
// Kernel 4: flash attention. Block = 4 waves, one (b,h,64-row q tile).
// Each wave owns 16 q rows. K/V tiles DMA'd into double-buffered LDS by the
// Tensor Data Mover (wave 0 issues, TENSORcnt pipelined one tile ahead);
// online softmax; P transposed through per-wave LDS staging for P@V WMMAs.
// ---------------------------------------------------------------------------
__global__ __launch_bounds__(128)
void attn_wmma(const unsigned short* __restrict__ Q,
               const unsigned short* __restrict__ Kb,
               const unsigned short* __restrict__ Vb,
               const int* __restrict__ pad,
               unsigned short* __restrict__ Y) {
  constexpr int T = 1024, H = 16, HD = 128;
  __shared__ unsigned short Ks[2][64][136];   // TDM-padded rows (256B + 16B)
  __shared__ unsigned short Vs[2][64][136];
  __shared__ unsigned short Ps[4][16][72];
  const int tid = threadIdx.x, lane = tid & 31, wid = tid >> 5;
  const int g = lane >> 4, lm = lane & 15;
  const int qt = blockIdx.x, h = blockIdx.y, b = blockIdx.z;
  const size_t headoff = (size_t)(b * H + h) * T * HD;
  const unsigned short* kbase = Kb + headoff;
  const unsigned short* vbase = Vb + headoff;

  const unsigned ks_off[2] = { (unsigned)(size_t)&Ks[0][0][0],
                               (unsigned)(size_t)&Ks[1][0][0] };
  const unsigned vs_off[2] = { (unsigned)(size_t)&Vs[0][0][0],
                               (unsigned)(size_t)&Vs[1][0][0] };

  // Q fragments: 16 rows x 128, split into 4 K=32 chunks.
  v16us qf[4];
  {
    const int r = qt * 64 + wid * 16 + lm;
    const unsigned short* qp = Q + headoff + (size_t)r * HD;
    #pragma unroll
    for (int kc = 0; kc < 4; ++kc)
      #pragma unroll
      for (int i = 0; i < 16; ++i) qf[kc][i] = qp[kc * 32 + kmap(i, g)];
  }

  v8f yacc[8];
  #pragma unroll
  for (int sn = 0; sn < 8; ++sn)
    #pragma unroll
    for (int e = 0; e < 8; ++e) yacc[sn][e] = 0.0f;
  float mrow[8], lrow[8];
  #pragma unroll
  for (int e = 0; e < 8; ++e) { mrow[e] = -3.0e38f; lrow[e] = 0.0f; }
  const float scale = 0.08838834764831845f;   // 1/sqrt(128)

  // prologue: kick off tile 0 DMA (wave 0 only; TDM ignores EXEC, one DMA/wave)
  if (wid == 0) {
    tdm_load_tile(ks_off[0], kbase);
    tdm_load_tile(vs_off[0], vbase);
  }

  for (int kt = 0; kt <= qt; ++kt) {
    const int cur = kt & 1, nxt = cur ^ 1;
    if (wid == 0) {
      if (kt < qt) {   // prefetch next tile, then wait for current (in-order)
        tdm_load_tile(ks_off[nxt], kbase + (size_t)(kt + 1) * 64 * HD);
        tdm_load_tile(vs_off[nxt], vbase + (size_t)(kt + 1) * 64 * HD);
        __builtin_amdgcn_s_wait_tensorcnt(2);
      } else {
        __builtin_amdgcn_s_wait_tensorcnt(0);
      }
    }
    __syncthreads();

    // scores: 16x64 per wave (4 WMMA x 4 K-chunks)
    float sreg[4][8];
    #pragma unroll
    for (int sn = 0; sn < 4; ++sn) {
      v8f s;
      #pragma unroll
      for (int e = 0; e < 8; ++e) s[e] = 0.0f;
      const int n = sn * 16 + lm;
      #pragma unroll
      for (int kc = 0; kc < 4; ++kc) {
        v16us kf;
        #pragma unroll
        for (int i = 0; i < 16; ++i) kf[i] = Ks[cur][n][kc * 32 + kmap(i, g)];
        s = wmma_bf16(qf[kc], kf, s);
      }
      const int col = kt * 64 + sn * 16 + lm;
      const int pm = pad[b * T + col];
      #pragma unroll
      for (int e = 0; e < 8; ++e) {
        const int row = qt * 64 + wid * 16 + g * 8 + e;
        const bool keep = (col <= row) && (pm != 0);
        sreg[sn][e] = keep ? s[e] * scale : -1.0e30f;
      }
    }

    // online softmax (row stats reduced across the 16-lane half-groups)
    float alpha[8];
    #pragma unroll
    for (int e = 0; e < 8; ++e) {
      float m = fmaxf(fmaxf(sreg[0][e], sreg[1][e]), fmaxf(sreg[2][e], sreg[3][e]));
      #pragma unroll
      for (int off = 8; off >= 1; off >>= 1) m = fmaxf(m, __shfl_xor(m, off, 16));
      const float mn = fmaxf(mrow[e], m);
      alpha[e] = __expf(mrow[e] - mn);
      mrow[e] = mn;
      float ps = 0.0f;
      #pragma unroll
      for (int sn = 0; sn < 4; ++sn) {
        const float p = __expf(sreg[sn][e] - mn);
        ps += p;
        Ps[wid][g * 8 + e][sn * 16 + lm] = f2bf(p);   // C-layout -> row-major
      }
      #pragma unroll
      for (int off = 8; off >= 1; off >>= 1) ps += __shfl_xor(ps, off, 16);
      lrow[e] = lrow[e] * alpha[e] + ps;
    }
    #pragma unroll
    for (int sn = 0; sn < 8; ++sn)
      #pragma unroll
      for (int e = 0; e < 8; ++e) yacc[sn][e] *= alpha[e];

    // make the per-wave LDS P staging visible before the transposed reads
    asm volatile("s_wait_dscnt 0" ::: "memory");

    // y += P(16x64) @ V(64x128): 2 K-chunks x 8 N-subtiles
    #pragma unroll
    for (int kc = 0; kc < 2; ++kc) {
      v16us pf;
      #pragma unroll
      for (int i = 0; i < 16; ++i) pf[i] = Ps[wid][lm][kc * 32 + kmap(i, g)];
      #pragma unroll
      for (int sn = 0; sn < 8; ++sn) {
        v16us vf;
        const int n = sn * 16 + lm;
        #pragma unroll
        for (int i = 0; i < 16; ++i) vf[i] = Vs[cur][kc * 32 + kmap(i, g)][n];
        yacc[sn] = wmma_bf16(pf, vf, yacc[sn]);
      }
    }
    __syncthreads();
  }

  // finalize: divide by l, write bf16 into (B,T,C) layout
  #pragma unroll
  for (int e = 0; e < 8; ++e) {
    const float inv = 1.0f / lrow[e];
    const int row = qt * 64 + wid * 16 + g * 8 + e;
    const size_t ybase = ((size_t)(b * T + row)) * (H * HD) + h * HD;
    #pragma unroll
    for (int sn = 0; sn < 8; ++sn)
      Y[ybase + sn * 16 + lm] = f2bf(yacc[sn][e] * inv);
  }
}

// ---------------------------------------------------------------------------
extern "C" void kernel_launch(void* const* d_in, const int* in_sizes, int n_in,
                              void* d_out, int out_size, void* d_ws, size_t ws_size,
                              hipStream_t stream) {
  (void)in_sizes; (void)n_in; (void)out_size; (void)ws_size;
  const float* x   = (const float*)d_in[0];
  const float* cs  = (const float*)d_in[1];
  const int*   tok = (const int*)d_in[2];
  const int*   pad = (const int*)d_in[3];
  const float* Wa  = (const float*)d_in[4];
  const float* Wp  = (const float*)d_in[5];
  float* out = (float*)d_out;

  // workspace carve-up (~66 MB)
  char* w = (char*)d_ws;
  float* sinb = (float*)w;            w += (size_t)4 * 1024 * 64 * 4;
  float* cosb = (float*)w;            w += (size_t)4 * 1024 * 64 * 4;
  unsigned short* Q  = (unsigned short*)w; w += (size_t)4 * 16 * 1024 * 128 * 2;
  unsigned short* Kq = (unsigned short*)w; w += (size_t)4 * 16 * 1024 * 128 * 2;
  unsigned short* V  = (unsigned short*)w; w += (size_t)4 * 16 * 1024 * 128 * 2;
  unsigned short* Y  = (unsigned short*)w; w += (size_t)4 * 16 * 1024 * 128 * 2;

  // 1) rotary time base -> sin/cos tables
  sincos_kernel<<<4, 1024, 0, stream>>>(tok, sinb, cosb);
  // 2) qkv = x @ W_attn, scattered to head-layout bf16 Q/K/V
  gemm_wmma<1, float><<<dim3(48, 32), 256, 0, stream>>>(
      x, Wa, nullptr, Q, Kq, V, 4096, 6144, 2048);
  // 3) rope + overrides + v scaling
  rope_kernel<<<16384, 256, 0, stream>>>(Q, Kq, V, sinb, cosb, cs);
  // 4) flash attention -> Y (B,T,C) bf16
  attn_wmma<<<dim3(16, 16, 4), 128, 0, stream>>>(Q, Kq, V, pad, Y);
  // 5) out = Y @ W_proj (f32 output)
  gemm_wmma<0, unsigned short><<<dim3(16, 32), 256, 0, stream>>>(
      Y, Wp, out, nullptr, nullptr, nullptr, 4096, 2048, 2048);
}
